// SpatialMambaBlock_9225589752274
// MI455X (gfx1250) — matrix operand order repaired and checked by reading, not compiled, not run on hardware
//
#include <hip/hip_runtime.h>
#include <math.h>

// ---------------------------------------------------------------------------
// Types for CDNA5 WMMA (wave32)
// ---------------------------------------------------------------------------
typedef __attribute__((ext_vector_type(16))) __bf16 v16bf;
typedef __attribute__((ext_vector_type(8)))  float  v8f;

union Frag16 { unsigned int u[8]; v16bf v; };

__device__ __forceinline__ unsigned short f2bf(float f) {
  __bf16 h = (__bf16)f;                       // RNE convert
  return __builtin_bit_cast(unsigned short, h);
}

// Model dims (compile-time constants from the reference)
#define DIMC     256
#define D_STATE  16
#define D_CONV   4
#define D_INNER  512
#define DT_RANK  16
#define MLP_HID  1024
#define BB       2
#define TT       4
#define HH       32
#define WW       32
#define NTOK     1024                 // H*W per (b,t)
#define NSEQ     8                    // B*T sequences
#define MTOK     8192                 // total tokens

// ---------------------------------------------------------------------------
// fp32 -> bf16 (bit-stored as ushort) weight conversion
// ---------------------------------------------------------------------------
__global__ void k_f32_to_bf16(const float* __restrict__ src,
                              unsigned short* __restrict__ dst, int n) {
  int i = blockIdx.x * blockDim.x + threadIdx.x;
  if (i < n) dst[i] = f2bf(src[i]);
}

// ---------------------------------------------------------------------------
// Pack (B,T,C,H,W) -> token-major, LayerNorm1, emit row-order and col-order
// bf16 activations plus fp32 shortcut.
// ---------------------------------------------------------------------------
__global__ __launch_bounds__(256)
void k_pack_ln(const float* __restrict__ x_in,
               const float* __restrict__ lnw, const float* __restrict__ lnb,
               float* __restrict__ shortcut,
               unsigned short* __restrict__ xn_row,
               unsigned short* __restrict__ xn_col) {
  const int tok = blockIdx.x;            // 0..8191
  const int c   = threadIdx.x;           // 0..255
  const int bt  = tok >> 10;
  const int n   = tok & 1023;
  const int h   = n >> 5;
  const int w   = n & 31;
  float v = x_in[((size_t)bt * DIMC + c) * NTOK + n];
  shortcut[(size_t)tok * DIMC + c] = v;

  __shared__ float red[256];
  red[c] = v; __syncthreads();
  for (int s = 128; s > 0; s >>= 1) { if (c < s) red[c] += red[c + s]; __syncthreads(); }
  float mu = red[0] * (1.0f / 256.0f);
  __syncthreads();
  float d = v - mu;
  red[c] = d * d; __syncthreads();
  for (int s = 128; s > 0; s >>= 1) { if (c < s) red[c] += red[c + s]; __syncthreads(); }
  float var = red[0] * (1.0f / 256.0f);
  float y = d * rsqrtf(var + 1e-5f) * lnw[c] + lnb[c];
  unsigned short yb = f2bf(y);
  xn_row[(size_t)tok * DIMC + c] = yb;
  int tokv = (bt << 10) + (w << 5) + h;       // v-order: w-major
  xn_col[(size_t)tokv * DIMC + c] = yb;
}

// ---------------------------------------------------------------------------
// Wide WMMA GEMM: Out[M,N](f32) = A[M,K](bf16) * W[N,K]^T (+bias)(+=)
// Each wave owns a 16x64 tile (4 accumulators); 8 waves/block cover 128 rows
// sharing one 64-column W stripe. The W stripe's 32-K chunk (4 KB) is staged
// into LDS per K-step with CDNA5 async loads (ASYNCcnt path), so W global
// traffic is amortized across all 8 waves.
// flags: bit0 accumulate into Out, bit1 add bias[col].
// Requires N % 64 == 0, K % 32 == 0, M % 128 == 0 (true for all call sites).
// ---------------------------------------------------------------------------
__global__ __launch_bounds__(256)
void k_gemm_bf16_w64(const unsigned short* __restrict__ Abf,
                     const unsigned short* __restrict__ Wbf,
                     const float* __restrict__ bias,
                     float* __restrict__ Out,
                     int M, int N, int K, int flags) {
  __shared__ unsigned int lb[1024];            // 64 cols x 16 uints (4 KB)

  const int lane  = threadIdx.x & 31;
  const int wave  = threadIdx.x >> 5;
  const int row0  = (blockIdx.y * 8 + wave) * 16;
  const int col0  = blockIdx.x * 64;

  const int l15 = lane & 15;
  const int kg  = lane >> 4;                   // 0 or 1

  // Cooperative W staging: thread t loads 16B of row (col0 + t/4), segment t%4.
  const int sc  = threadIdx.x >> 2;            // 0..63  (column within stripe)
  const int seg = threadIdx.x & 3;             // 0..3   (16B segment of 64B row)
  const unsigned short* gW = Wbf + (size_t)(col0 + sc) * K + seg * 8;
  const unsigned int lds_dst =
      (unsigned int)(size_t)lb + (unsigned int)((sc * 16 + seg * 4) * 4);

  const unsigned int* Arow =
      reinterpret_cast<const unsigned int*>(Abf + (size_t)(row0 + l15) * K);

  v8f acc0 = {}, acc1 = {}, acc2 = {}, acc3 = {};
  for (int k0 = 0; k0 < K; k0 += 32) {
    // ---- async-stage W[col0..col0+63][k0..k0+31] into LDS ----
    unsigned long long ga = (unsigned long long)(size_t)(gW + k0);
    asm volatile("global_load_async_to_lds_b128 %0, %1, off"
                 :: "v"(lds_dst), "v"(ga) : "memory");
    asm volatile("s_wait_asynccnt 0x0" ::: "memory");
    __syncthreads();

    // ---- A fragment (direct, b128-coalesced) ----
    Frag16 fa;
#pragma unroll
    for (int j = 0; j < 8; ++j) {
      int ka = k0 + ((j < 4) ? 0 : 16) + kg * 8 + (j & 3) * 2;
      fa.u[j] = Arow[ka >> 1];
    }
    if (k0 + 32 < K) __builtin_prefetch(&Arow[(k0 + 32) >> 1], 0, 0);

    // ---- 4 B fragments from LDS, 4 WMMAs reusing the A fragment ----
#pragma unroll
    for (int nsub = 0; nsub < 4; ++nsub) {
      Frag16 fb;
      const int c = nsub * 16 + l15;
#pragma unroll
      for (int j = 0; j < 8; ++j) fb.u[j] = lb[c * 16 + ((j < 4) ? 0 : 8) + kg * 4 + (j & 3)];
      v8f* accp = (nsub == 0) ? &acc0 : (nsub == 1) ? &acc1
                : (nsub == 2) ? &acc2 : &acc3;
      *accp = __builtin_amdgcn_wmma_f32_16x16x32_bf16(false, fa.v, false, fb.v,
                                                      (short)0, *accp, false, false);
    }
    __syncthreads();                            // protect LDS before next stage
  }

  v8f* accs[4] = {&acc0, &acc1, &acc2, &acc3};
#pragma unroll
  for (int nsub = 0; nsub < 4; ++nsub) {
    const int colOut = col0 + nsub * 16 + l15;
    const float bval = (flags & 2) ? bias[colOut] : 0.0f;
#pragma unroll
    for (int j = 0; j < 8; ++j) {
      int rowOut = row0 + kg * 8 + j;           // C/D: VGPR j -> M = j (+8 hi lanes)
      size_t idx = (size_t)rowOut * N + colOut;
      float v = (*accs[nsub])[j] + bval;
      if (flags & 1) v += Out[idx];
      Out[idx] = v;
    }
  }
}

// ---------------------------------------------------------------------------
// Narrow WMMA GEMM (one 16x16 tile per wave) — used for N=48 (x_proj).
// ---------------------------------------------------------------------------
__global__ __launch_bounds__(256)
void k_gemm_bf16(const unsigned short* __restrict__ Abf,
                 const unsigned short* __restrict__ Wbf,
                 const float* __restrict__ bias,
                 float* __restrict__ Out,
                 int M, int N, int K, int flags) {
  const int lane  = threadIdx.x & 31;
  const int wave  = threadIdx.x >> 5;
  const int mTile = blockIdx.y * 8 + wave;
  const int nTile = blockIdx.x;
  const int row0  = mTile * 16;
  const int col0  = nTile * 16;
  if (row0 >= M) return;

  const int l15 = lane & 15;
  const int kg  = lane >> 4;
  const unsigned int* Arow =
      reinterpret_cast<const unsigned int*>(Abf + (size_t)(row0 + l15) * K);
  const unsigned int* Wrow =
      reinterpret_cast<const unsigned int*>(Wbf + (size_t)(col0 + l15) * K);

  v8f acc = {};
  for (int k0 = 0; k0 < K; k0 += 32) {
    Frag16 fa, fb;
#pragma unroll
    for (int j = 0; j < 8; ++j) {
      int ka = k0 + ((j < 4) ? 0 : 16) + kg * 8 + (j & 3) * 2;
      fa.u[j] = Arow[ka >> 1];
      int kb = k0 + ((j < 4) ? 0 : 16) + kg * 8 + (j & 3) * 2;
      fb.u[j] = Wrow[kb >> 1];
    }
    acc = __builtin_amdgcn_wmma_f32_16x16x32_bf16(false, fa.v, false, fb.v,
                                                  (short)0, acc, false, false);
  }

  const int colOut = col0 + l15;
  const float bval = (flags & 2) ? bias[colOut] : 0.0f;
#pragma unroll
  for (int j = 0; j < 8; ++j) {
    int rowOut = row0 + kg * 8 + j;
    size_t idx = (size_t)rowOut * N + colOut;
    float v = acc[j] + bval;
    if (flags & 1) v += Out[idx];
    Out[idx] = v;
  }
}

// ---------------------------------------------------------------------------
// Direction-aware causal depthwise conv1d (taps=4) + bias + SiLU.
// ---------------------------------------------------------------------------
__global__ __launch_bounds__(256)
void k_conv_silu(const float* __restrict__ xz,        // [tok][1024], ch 0..511 = xin
                 const float* __restrict__ cw,        // [512][4]
                 const float* __restrict__ cb, int dir,
                 float* __restrict__ xin_f,
                 unsigned short* __restrict__ xin_h) {
  int idx = blockIdx.x * 256 + threadIdx.x;           // tok*512 + d
  int d   = idx & 511;
  int tok = idx >> 9;
  int s   = tok >> 10;
  int l   = tok & 1023;
  float acc = cb[d];
#pragma unroll
  for (int k = 0; k < 4; ++k) {
    int lp = dir ? (l + 3 - k) : (l - 3 + k);
    if (lp >= 0 && lp < NTOK)
      acc += cw[d * 4 + k] * xz[(size_t)(s * NTOK + lp) * 1024 + d];
  }
  float y = acc / (1.0f + __expf(-acc));              // silu
  xin_f[(size_t)tok * D_INNER + d] = y;
  xin_h[(size_t)tok * D_INNER + d] = f2bf(y);
}

// ---------------------------------------------------------------------------
// dt = softplus(dbc[:, :16] @ dt_proj_w^T + dt_proj_b)   (K=16, VALU)
// ---------------------------------------------------------------------------
__global__ __launch_bounds__(256)
void k_dtproj(const float* __restrict__ dbc,          // [tok][48]
              const float* __restrict__ w,            // [512][16]
              const float* __restrict__ b,
              float* __restrict__ dt) {
  int idx = blockIdx.x * 256 + threadIdx.x;
  int d   = idx & 511;
  int tok = idx >> 9;
  float acc = b[d];
  const float* r = dbc + (size_t)tok * 48;
#pragma unroll
  for (int k = 0; k < DT_RANK; ++k) acc += r[k] * w[d * DT_RANK + k];
  dt[(size_t)tok * D_INNER + d] = (acc > 20.0f) ? acc : log1pf(__expf(acc));
}

// ---------------------------------------------------------------------------
// Selective scan (sequential over L=1024) fused with +xin*D and *silu(z).
// ---------------------------------------------------------------------------
__global__ __launch_bounds__(256)
void k_scan(const float* __restrict__ dt,             // [tok][512]
            const float* __restrict__ xin_f,          // [tok][512]
            const float* __restrict__ dbc,            // [tok][48] (B at 16, C at 32)
            const float* __restrict__ xz,             // [tok][1024], z at ch 512+
            const float* __restrict__ A_log,          // [512][16]
            const float* __restrict__ Dp,             // [512]
            int dir,
            unsigned short* __restrict__ yact) {      // [tok][512] bf16
  int s = blockIdx.x >> 1;
  int d = ((blockIdx.x & 1) << 8) + threadIdx.x;
  float A[D_STATE], hst[D_STATE];
#pragma unroll
  for (int j = 0; j < D_STATE; ++j) {
    A[j]   = -__expf(A_log[d * D_STATE + j]);
    hst[j] = 0.0f;
  }
  float Dv = Dp[d];
  for (int i = 0; i < NTOK; ++i) {
    int l = dir ? (NTOK - 1 - i) : i;
    size_t tok = (size_t)s * NTOK + l;
    float dtv = dt[tok * D_INNER + d];
    float xv  = xin_f[tok * D_INNER + d];
    const float* bc = dbc + tok * 48;
    float y = 0.0f;
    float dx = dtv * xv;
#pragma unroll
    for (int j = 0; j < D_STATE; ++j) {
      hst[j] = __expf(dtv * A[j]) * hst[j] + dx * bc[16 + j];
      y += hst[j] * bc[32 + j];
    }
    y += xv * Dv;
    float zv = xz[tok * 1024 + D_INNER + d];
    float o  = y * (zv / (1.0f + __expf(-zv)));
    yact[tok * D_INNER + d] = f2bf(o);
  }
}

// ---------------------------------------------------------------------------
// Fusion gate: ctx = mean_tokens(0.5*(out_h+out_v)) ; g = sigmoid(g2(relu(g1 ctx)))
// ---------------------------------------------------------------------------
__global__ __launch_bounds__(256)
void k_gate(const float* __restrict__ out_h, const float* __restrict__ out_v,
            const float* __restrict__ g1w, const float* __restrict__ g1b,
            const float* __restrict__ g2w, const float* __restrict__ g2b,
            float* __restrict__ g) {
  int b = blockIdx.x, c = threadIdx.x;
  float sum = 0.0f;
  size_t base = (size_t)b * (TT * NTOK) * DIMC;
  for (int t = 0; t < TT * NTOK; ++t) {
    size_t i = base + (size_t)t * DIMC + c;
    sum += out_h[i] + out_v[i];                        // order-independent sum
  }
  __shared__ float ctx[DIMC];
  __shared__ float hid[64];
  ctx[c] = sum * 0.5f / (float)(TT * NTOK);
  __syncthreads();
  if (c < 64) {
    float a = g1b[c];
    for (int k = 0; k < DIMC; ++k) a += g1w[c * DIMC + k] * ctx[k];
    hid[c] = fmaxf(a, 0.0f);
  }
  __syncthreads();
  float a = g2b[c];
  for (int k = 0; k < 64; ++k) a += g2w[c * 64 + k] * hid[k];
  g[b * DIMC + c] = 1.0f / (1.0f + __expf(-a));
}

// ---------------------------------------------------------------------------
// Fuse residual + gated scans, then LayerNorm2 -> bf16 for fc1 GEMM.
// ---------------------------------------------------------------------------
__global__ __launch_bounds__(256)
void k_fuse_ln2(const float* __restrict__ shortcut,
                const float* __restrict__ out_h,
                const float* __restrict__ out_v,     // stored in v-order
                const float* __restrict__ g,
                const float* __restrict__ lnw, const float* __restrict__ lnb,
                float* __restrict__ x2, unsigned short* __restrict__ xn2) {
  int tok = blockIdx.x, c = threadIdx.x;
  int bt = tok >> 10, n = tok & 1023;
  int h = n >> 5, w = n & 31;
  int tokv = (bt << 10) + (w << 5) + h;
  int b = bt >> 2;                                    // T = 4
  float gv = g[b * DIMC + c];
  float v = shortcut[(size_t)tok * DIMC + c]
          + gv * out_h[(size_t)tok * DIMC + c]
          + (1.0f - gv) * out_v[(size_t)tokv * DIMC + c];
  x2[(size_t)tok * DIMC + c] = v;

  __shared__ float red[256];
  red[c] = v; __syncthreads();
  for (int s = 128; s > 0; s >>= 1) { if (c < s) red[c] += red[c + s]; __syncthreads(); }
  float mu = red[0] * (1.0f / 256.0f);
  __syncthreads();
  float d = v - mu;
  red[c] = d * d; __syncthreads();
  for (int s = 128; s > 0; s >>= 1) { if (c < s) red[c] += red[c + s]; __syncthreads(); }
  float var = red[0] * (1.0f / 256.0f);
  xn2[(size_t)tok * DIMC + c] = f2bf(d * rsqrtf(var + 1e-5f) * lnw[c] + lnb[c]);
}

// ---------------------------------------------------------------------------
// 3x3 depthwise conv (SAME) + bias + exact GELU  -> bf16 for fc2 GEMM.
// ---------------------------------------------------------------------------
__global__ __launch_bounds__(256)
void k_dwconv_gelu(const float* __restrict__ hdd,    // [tok][1024]
                   const float* __restrict__ dw,     // [3][3][1][1024]
                   const float* __restrict__ db,
                   unsigned short* __restrict__ h2) {
  int idx = blockIdx.x * 256 + threadIdx.x;          // tok*1024 + ch
  int ch  = idx & 1023;
  int tok = idx >> 10;
  int bt  = tok >> 10, n = tok & 1023;
  int h   = n >> 5, w = n & 31;
  float acc = db[ch];
#pragma unroll
  for (int kh = 0; kh < 3; ++kh) {
    int hh = h + kh - 1;
    if (hh < 0 || hh >= HH) continue;
#pragma unroll
    for (int kw = 0; kw < 3; ++kw) {
      int ww = w + kw - 1;
      if (ww < 0 || ww >= WW) continue;
      size_t t2 = ((size_t)bt << 10) + (hh << 5) + ww;
      acc += dw[(kh * 3 + kw) * MLP_HID + ch] * hdd[t2 * MLP_HID + ch];
    }
  }
  float gl = 0.5f * acc * (1.0f + erff(acc * 0.70710678118f));
  h2[(size_t)tok * MLP_HID + ch] = f2bf(gl);
}

// ---------------------------------------------------------------------------
// Unpack token-major x2 back to (B,T,C,H,W).
// ---------------------------------------------------------------------------
__global__ __launch_bounds__(256)
void k_unpack(const float* __restrict__ x2, float* __restrict__ out) {
  int idx = blockIdx.x * 256 + threadIdx.x;
  int n = idx & 1023;
  int rest = idx >> 10;
  int c  = rest & 255;
  int bt = rest >> 8;
  out[idx] = x2[(((size_t)bt << 10) + n) * DIMC + c];
}

// ---------------------------------------------------------------------------
// Host launcher
// ---------------------------------------------------------------------------
enum {
  IN_X = 0,
  P_MH_INPROJ, P_MH_CONVW, P_MH_CONVB, P_MH_XPROJ, P_MH_DTW, P_MH_DTB,
  P_MH_ALOG, P_MH_D, P_MH_OUTPROJ,
  P_MV_INPROJ, P_MV_CONVW, P_MV_CONVB, P_MV_XPROJ, P_MV_DTW, P_MV_DTB,
  P_MV_ALOG, P_MV_D, P_MV_OUTPROJ,
  P_LN1W, P_LN1B, P_LN2W, P_LN2B,
  P_G1W, P_G1B, P_G2W, P_G2B,
  P_FC1W, P_FC1B, P_DWW, P_DWB, P_FC2W, P_FC2B
};

extern "C" void kernel_launch(void* const* d_in, const int* in_sizes, int n_in,
                              void* d_out, int out_size, void* d_ws, size_t ws_size,
                              hipStream_t stream) {
  (void)in_sizes; (void)n_in; (void)out_size; (void)ws_size;
  const float* x_in = (const float*)d_in[IN_X];
  auto F = [&](int i) { return (const float*)d_in[i]; };

  // ---- workspace bump allocation (256B aligned) ----
  char* ws = (char*)d_ws;
  size_t off = 0;
  auto alloc = [&](size_t bytes) -> char* {
    char* p = ws + off;
    off += (bytes + 255) & ~(size_t)255;
    return p;
  };
  float*          shortcut = (float*)alloc((size_t)MTOK * DIMC * 4);
  unsigned short* xn_row   = (unsigned short*)alloc((size_t)MTOK * DIMC * 2);
  unsigned short* xn_col   = (unsigned short*)alloc((size_t)MTOK * DIMC * 2);
  float*          out_h    = (float*)alloc((size_t)MTOK * DIMC * 4);
  float*          out_v    = (float*)alloc((size_t)MTOK * DIMC * 4);
  float*          x2       = (float*)alloc((size_t)MTOK * DIMC * 4);
  unsigned short* xn2      = (unsigned short*)alloc((size_t)MTOK * DIMC * 2);
  float*          gbuf     = (float*)alloc(BB * DIMC * 4);
  unsigned short* wb_in[2]  = {(unsigned short*)alloc(2 * D_INNER * DIMC * 2),
                               (unsigned short*)alloc(2 * D_INNER * DIMC * 2)};
  unsigned short* wb_xp[2]  = {(unsigned short*)alloc(48 * D_INNER * 2),
                               (unsigned short*)alloc(48 * D_INNER * 2)};
  unsigned short* wb_op[2]  = {(unsigned short*)alloc(DIMC * D_INNER * 2),
                               (unsigned short*)alloc(DIMC * D_INNER * 2)};
  unsigned short* wb_fc1   = (unsigned short*)alloc(MLP_HID * DIMC * 2);
  unsigned short* wb_fc2   = (unsigned short*)alloc(DIMC * MLP_HID * 2);
  float*          xz       = (float*)alloc((size_t)MTOK * 1024 * 4);   // reused as hdd
  float*          xin_f    = (float*)alloc((size_t)MTOK * D_INNER * 4);// reused as h2 (bf16)
  unsigned short* xin_h    = (unsigned short*)alloc((size_t)MTOK * D_INNER * 2);
  float*          dbc      = (float*)alloc((size_t)MTOK * 48 * 4);
  float*          dtb      = (float*)alloc((size_t)MTOK * D_INNER * 4);
  unsigned short* yact     = (unsigned short*)alloc((size_t)MTOK * D_INNER * 2);
  float*          hdd      = xz;                       // alias (xz dead by then)
  unsigned short* h2       = (unsigned short*)xin_f;   // alias (xin_f dead by then)

  // ---- weight conversions to bf16 ----
  auto conv_w = [&](const float* src, unsigned short* dst, int n) {
    k_f32_to_bf16<<<(n + 255) / 256, 256, 0, stream>>>(src, dst, n);
  };
  conv_w(F(P_MH_INPROJ), wb_in[0], 2 * D_INNER * DIMC);
  conv_w(F(P_MV_INPROJ), wb_in[1], 2 * D_INNER * DIMC);
  conv_w(F(P_MH_XPROJ),  wb_xp[0], 48 * D_INNER);
  conv_w(F(P_MV_XPROJ),  wb_xp[1], 48 * D_INNER);
  conv_w(F(P_MH_OUTPROJ), wb_op[0], DIMC * D_INNER);
  conv_w(F(P_MV_OUTPROJ), wb_op[1], DIMC * D_INNER);
  conv_w(F(P_FC1W), wb_fc1, MLP_HID * DIMC);
  conv_w(F(P_FC2W), wb_fc2, DIMC * MLP_HID);

  // ---- pack + LN1 ----
  k_pack_ln<<<MTOK, 256, 0, stream>>>(x_in, F(P_LN1W), F(P_LN1B),
                                      shortcut, xn_row, xn_col);

  auto gemm = [&](const unsigned short* A, const unsigned short* W,
                  const float* bias, float* Out, int M, int N, int K, int flags) {
    if ((N & 63) == 0) {
      dim3 grid(N / 64, M / 128);
      k_gemm_bf16_w64<<<grid, 256, 0, stream>>>(A, W, bias, Out, M, N, K, flags);
    } else {
      dim3 grid(N / 16, (M + 127) / 128);
      k_gemm_bf16<<<grid, 256, 0, stream>>>(A, W, bias, Out, M, N, K, flags);
    }
  };

  // ---- 2 mamba modules (mh then mv), each bidirectional ----
  const int pbase[2] = {P_MH_INPROJ, P_MV_INPROJ};
  const unsigned short* xn_m[2] = {xn_row, xn_col};
  float* out_m[2] = {out_h, out_v};
  for (int m = 0; m < 2; ++m) {
    int pb = pbase[m];
    // in_proj: xz[tok][1024] = xn @ in_proj_w^T   (direction-independent)
    gemm(xn_m[m], wb_in[m], nullptr, xz, MTOK, 2 * D_INNER, DIMC, 0);
    for (int dir = 0; dir < 2; ++dir) {
      k_conv_silu<<<(MTOK * D_INNER) / 256, 256, 0, stream>>>(
          xz, F(pb + 1), F(pb + 2), dir, xin_f, xin_h);
      // x_proj: dbc = xin @ x_proj_w^T  (N=48 -> narrow kernel)
      gemm(xin_h, wb_xp[m], nullptr, dbc, MTOK, 48, D_INNER, 0);
      k_dtproj<<<(MTOK * D_INNER) / 256, 256, 0, stream>>>(
          dbc, F(pb + 4), F(pb + 5), dtb);
      k_scan<<<NSEQ * 2, 256, 0, stream>>>(
          dtb, xin_f, dbc, xz, F(pb + 6), F(pb + 7), dir, yact);
      // out_proj: out_m (+)= yact @ out_proj_w^T
      gemm(yact, wb_op[m], nullptr, out_m[m], MTOK, DIMC, D_INNER, dir ? 1 : 0);
    }
  }

  // ---- fusion gate + fuse + LN2 ----
  k_gate<<<BB, 256, 0, stream>>>(out_h, out_v, F(P_G1W), F(P_G1B),
                                 F(P_G2W), F(P_G2B), gbuf);
  k_fuse_ln2<<<MTOK, 256, 0, stream>>>(shortcut, out_h, out_v, gbuf,
                                       F(P_LN2W), F(P_LN2B), x2, xn2);

  // ---- MLP: fc1 -> dwconv+gelu -> fc2 (accumulate into x2) ----
  gemm(xn2, wb_fc1, F(P_FC1B), hdd, MTOK, MLP_HID, DIMC, 2);
  k_dwconv_gelu<<<(MTOK * MLP_HID) / 256, 256, 0, stream>>>(
      hdd, F(P_DWW), F(P_DWB), h2);
  gemm(h2, wb_fc2, F(P_FC2B), x2, MTOK, DIMC, MLP_HID, 3);

  // ---- unpack to (B,T,C,H,W) ----
  k_unpack<<<(MTOK * DIMC) / 256, 256, 0, stream>>>(x2, (float*)d_out);
}
